// MambaBlock_4501125726258
// MI455X (gfx1250) — compile-verified, matrix operand order checked by reading, not run on hardware
//
#include <hip/hip_runtime.h>

// ---------------------------------------------------------------------------
// Mamba block (3 branches: fwd, reverse, slice-permuted) for gfx1250 / MI455X.
// GEMMs use v_wmma_f32_16x16x32_f16 (wave32, f32 accumulate), with branch-free
// unrolled inner loops (K templated; tail folded at compile time).
// Selective scan uses a 3-pass chunked linear-recurrence scan for parallelism.
// ---------------------------------------------------------------------------

typedef __attribute__((ext_vector_type(16))) _Float16 v16h;
typedef __attribute__((ext_vector_type(8)))  float    v8f;

// Problem constants
constexpr int BATCH = 2;
constexpr int SEQ   = 2560;
constexpr int DM    = 768;        // d_model
constexpr int DI    = 1536;       // d_inner
constexpr int DS    = 16;         // d_state
constexpr int DR    = 48;         // dt_rank
constexpr int NX    = DR + 2*DS;  // 80  (x_proj output dim)
constexpr int NE    = 2 * DI;     // 3072 (in_proj output dim)
constexpr int BL    = BATCH * SEQ;   // 5120 rows for all GEMMs
constexpr int NSL   = 5;
constexpr int LS    = SEQ / NSL;     // 512
constexpr int CH    = 20;            // scan chunks
constexpr int CL    = SEQ / CH;      // 128 steps per chunk

__device__ __forceinline__ int physmap(int j, int br) {
  if (br == 0) return j;
  if (br == 1) return SEQ - 1 - j;
  int s = j % NSL, t = j / NSL;     // slice branch: xz_s[:, j] = xz[:, s*LS + t]
  return s * LS + t;
}

__device__ __forceinline__ float siluf(float x) {
  return x / (1.0f + __expf(-x));
}

// --- WMMA operand packing ---------------------------------------------------
// ISA 16-bit A/B 16x32 layout: lane holds row/col (lane&15); element i covers
// k = (i>>3)*16 + (lane>>4)*8 + (i&7). So each lane loads two contiguous
// 8-float runs at p (= row + k0 + hi*8) and p+16. All loads aligned float4s.

__device__ __forceinline__ v16h load_full(const float* __restrict__ p) {
  float4 f0 = *reinterpret_cast<const float4*>(p);
  float4 f1 = *reinterpret_cast<const float4*>(p + 4);
  float4 f2 = *reinterpret_cast<const float4*>(p + 16);
  float4 f3 = *reinterpret_cast<const float4*>(p + 20);
  v16h v;
  v[0]  = (_Float16)f0.x; v[1]  = (_Float16)f0.y; v[2]  = (_Float16)f0.z; v[3]  = (_Float16)f0.w;
  v[4]  = (_Float16)f1.x; v[5]  = (_Float16)f1.y; v[6]  = (_Float16)f1.z; v[7]  = (_Float16)f1.w;
  v[8]  = (_Float16)f2.x; v[9]  = (_Float16)f2.y; v[10] = (_Float16)f2.z; v[11] = (_Float16)f2.w;
  v[12] = (_Float16)f3.x; v[13] = (_Float16)f3.y; v[14] = (_Float16)f3.z; v[15] = (_Float16)f3.w;
  return v;
}

// Tail group fill with COMPILE-TIME (wave-uniform) full/zero decision:
// group base is hi-independent; full iff base+16 <= K (covers both hi halves),
// all-zero iff base >= K. (Mixed case never occurs for K in {768,1536,48}.)
template<int K, int BASE, int E0>
__device__ __forceinline__ void fill_group(v16h& v, const float* __restrict__ row, int hi) {
  if constexpr (BASE + 16 <= K) {
    const float* p = row + BASE + hi * 8;
    float4 f0 = *reinterpret_cast<const float4*>(p);
    float4 f1 = *reinterpret_cast<const float4*>(p + 4);
    v[E0+0] = (_Float16)f0.x; v[E0+1] = (_Float16)f0.y;
    v[E0+2] = (_Float16)f0.z; v[E0+3] = (_Float16)f0.w;
    v[E0+4] = (_Float16)f1.x; v[E0+5] = (_Float16)f1.y;
    v[E0+6] = (_Float16)f1.z; v[E0+7] = (_Float16)f1.w;
  } else if constexpr (BASE >= K) {
#pragma unroll
    for (int j = 0; j < 8; ++j) v[E0+j] = (_Float16)0.0f;
  } else {
#pragma unroll
    for (int j = 0; j < 8; ++j) {
      int k = BASE + hi * 8 + j;
      v[E0+j] = (k < K) ? (_Float16)row[k] : (_Float16)0.0f;
    }
  }
}

template<int K, int K0>
__device__ __forceinline__ v16h load_tail(const float* __restrict__ row, int hi) {
  v16h v;
  fill_group<K, K0,      0>(v, row, hi);
  fill_group<K, K0 + 16, 8>(v, row, hi);
  return v;
}

// Generic GEMM: C[m][n] = sum_k A[m][k] * W[n][k]   (W row-major [N][K])
// ACT: 0 none, 1 bias + softplus (dt_proj)
// SMODE: 0 row-major C[m*ldc+n]; 1 "xz" transposed store C[b][n][l]
template<int K, int ACT, int SMODE>
__global__ __launch_bounds__(128)
void wmma_gemm_kernel(const float* __restrict__ A, int lda,
                      const float* __restrict__ W,
                      float* __restrict__ C,
                      const float* __restrict__ bias,
                      int N, int ldc) {
  const int lane = threadIdx.x & 31;
  const int wave = threadIdx.x >> 5;
  const int tn = blockIdx.x * 4 + wave;   // 4 waves per block, one 16x16 tile each
  const int tm = blockIdx.y;
  if (tn * 16 >= N) return;               // whole wave exits: EXEC all-ones for WMMA

  const int mrow = lane & 15;
  const int hi   = lane >> 4;
  const float* arow = A + (size_t)(tm * 16 + mrow) * lda;
  const float* wrow = W + (size_t)(tn * 16 + mrow) * K;

  v8f acc = {0.f, 0.f, 0.f, 0.f, 0.f, 0.f, 0.f, 0.f};
  constexpr int KF = K & ~31;             // full 32-wide chunks
#pragma unroll 8
  for (int k0 = 0; k0 < KF; k0 += 32) {
    v16h av = load_full(arow + k0 + hi * 8);
    v16h bv = load_full(wrow + k0 + hi * 8);
    acc = __builtin_amdgcn_wmma_f32_16x16x32_f16(
        false, av, false, bv, (short)0, acc, false, false);
  }
  if constexpr (KF < K) {                 // compile-time tail (K=48 only)
    v16h av = load_tail<K, KF>(arow, hi);
    v16h bv = load_tail<K, KF>(wrow, hi);
    acc = __builtin_amdgcn_wmma_f32_16x16x32_f16(
        false, av, false, bv, (short)0, acc, false, false);
  }

  const int ncol = tn * 16 + mrow;
  float bval = 0.0f;
  if constexpr (ACT == 1) bval = bias[ncol];
#pragma unroll
  for (int r = 0; r < 8; ++r) {
    int m = tm * 16 + r + hi * 8;         // C/D layout: VGPR r -> row r (+8 for hi lanes)
    float v = acc[r] + bval;
    if constexpr (ACT == 1) v = (v > 20.f) ? v : __logf(1.0f + __expf(v));
    if constexpr (SMODE == 0) {
      C[(size_t)m * ldc + ncol] = v;
    } else {                              // xz[b][e][l]
      int b = m / SEQ;
      int l = m - b * SEQ;
      C[(size_t)b * NE * SEQ + (size_t)ncol * SEQ + l] = v;
    }
  }
}

// Depthwise causal conv(4) + SiLU on the (possibly permuted) sequence, and
// precompute silu(z) in scan-friendly [b][j][d] layout.
__global__ void conv_silu_kernel(const float* __restrict__ xz,
                                 const float* __restrict__ cw,   // [DI][4]
                                 const float* __restrict__ cb,   // [DI]
                                 float* __restrict__ convout,    // [b][j][d]
                                 float* __restrict__ zout,       // [b][j][d]
                                 int branch) {
  int idx = blockIdx.x * blockDim.x + threadIdx.x;
  if (idx >= BATCH * SEQ * DI) return;
  int d = idx % DI;
  int j = (idx / DI) % SEQ;
  int b = idx / (DI * SEQ);
  const float* xrow = xz + ((size_t)b * NE + d) * SEQ;
  float acc = cb[d];
#pragma unroll
  for (int k = 0; k < 4; ++k) {
    int jj = j - 3 + k;
    if (jj >= 0) acc += cw[d * 4 + k] * xrow[physmap(jj, branch)];
  }
  convout[idx] = siluf(acc);
  float zv = xz[((size_t)b * NE + DI + d) * SEQ + physmap(j, branch)];
  zout[idx] = siluf(zv);
}

// Chunked linear scan, pass 1: per (b,d,chunk) compute chunk transfer function
// h_out = Aprod ⊙ h_in + Hend (run with h_in = 0).
__global__ void scan_pass1(const float* __restrict__ delta,    // [bl][DI] (softplus'ed)
                           const float* __restrict__ convout,  // [bl][DI]
                           const float* __restrict__ xdbl,     // [bl][NX]
                           const float* __restrict__ Alog,     // [DI][DS]
                           float* __restrict__ Aprod,          // [b][c][n][DI]
                           float* __restrict__ Hend) {
  int d = blockIdx.x * blockDim.x + threadIdx.x;
  int c = blockIdx.y, b = blockIdx.z;
  float An[DS], h[DS], ap[DS];
#pragma unroll
  for (int n = 0; n < DS; ++n) {
    An[n] = -__expf(Alog[d * DS + n]);
    h[n] = 0.f; ap[n] = 1.f;
  }
  for (int t = 0; t < CL; ++t) {
    int row = b * SEQ + c * CL + t;
    float dl = delta[(size_t)row * DI + d];
    float u  = convout[(size_t)row * DI + d];
    float du = dl * u;
    float Bv[DS];
    const float4* bp = reinterpret_cast<const float4*>(xdbl + (size_t)row * NX + DR);
#pragma unroll
    for (int q = 0; q < 4; ++q) reinterpret_cast<float4*>(Bv)[q] = bp[q];
#pragma unroll
    for (int n = 0; n < DS; ++n) {
      float e = __expf(dl * An[n]);
      h[n] = h[n] * e + du * Bv[n];
      ap[n] *= e;
    }
  }
  size_t base = (size_t)(b * CH + c) * DS * DI + d;
#pragma unroll
  for (int n = 0; n < DS; ++n) {
    Aprod[base + (size_t)n * DI] = ap[n];
    Hend [base + (size_t)n * DI] = h[n];
  }
}

// Pass 2: exclusive scan across the CH=20 chunks per (b,d,n).
__global__ void scan_pass2(const float* __restrict__ Aprod,
                           const float* __restrict__ Hend,
                           float* __restrict__ Hinit) {
  int idx = blockIdx.x * blockDim.x + threadIdx.x;
  if (idx >= BATCH * DS * DI) return;
  int d = idx % DI;
  int n = (idx / DI) & (DS - 1);
  int b = idx / (DI * DS);
  float h = 0.f;
  for (int c = 0; c < CH; ++c) {
    size_t o = (size_t)(b * CH + c) * DS * DI + (size_t)n * DI + d;
    Hinit[o] = h;
    h = Aprod[o] * h + Hend[o];
  }
}

// Pass 3: replay each chunk from its true initial state, emit
// y = (h·C + u*Dp) * silu(z), scattered back to physical order.
__global__ void scan_pass3(const float* __restrict__ delta,
                           const float* __restrict__ convout,
                           const float* __restrict__ zout,
                           const float* __restrict__ xdbl,
                           const float* __restrict__ Alog,
                           const float* __restrict__ Dp,
                           const float* __restrict__ Hinit,
                           float* __restrict__ ysum,           // [b][l][d]
                           int branch, int accum) {
  int d = blockIdx.x * blockDim.x + threadIdx.x;
  int c = blockIdx.y, b = blockIdx.z;
  float An[DS], h[DS];
  size_t hbase = (size_t)(b * CH + c) * DS * DI + d;
#pragma unroll
  for (int n = 0; n < DS; ++n) {
    An[n] = -__expf(Alog[d * DS + n]);
    h[n] = Hinit[hbase + (size_t)n * DI];
  }
  float Dv = Dp[d];
  for (int t = 0; t < CL; ++t) {
    int j = c * CL + t;
    int row = b * SEQ + j;
    float dl = delta[(size_t)row * DI + d];
    float u  = convout[(size_t)row * DI + d];
    float du = dl * u;
    float Bv[DS], Cv[DS];
    const float4* bp = reinterpret_cast<const float4*>(xdbl + (size_t)row * NX + DR);
    const float4* cp = reinterpret_cast<const float4*>(xdbl + (size_t)row * NX + DR + DS);
#pragma unroll
    for (int q = 0; q < 4; ++q) {
      reinterpret_cast<float4*>(Bv)[q] = bp[q];
      reinterpret_cast<float4*>(Cv)[q] = cp[q];
    }
    float y = 0.f;
#pragma unroll
    for (int n = 0; n < DS; ++n) {
      float e = __expf(dl * An[n]);
      h[n] = h[n] * e + du * Bv[n];
      y += h[n] * Cv[n];
    }
    float val = (y + u * Dv) * zout[(size_t)row * DI + d];
    int l = physmap(j, branch);
    size_t o = ((size_t)b * SEQ + l) * DI + d;
    if (accum) ysum[o] += val; else ysum[o] = val;
  }
}

// ---------------------------------------------------------------------------

extern "C" void kernel_launch(void* const* d_in, const int* in_sizes, int n_in,
                              void* d_out, int out_size, void* d_ws, size_t ws_size,
                              hipStream_t stream) {
  const float* hidden     = (const float*)d_in[0];
  const float* in_proj_w  = (const float*)d_in[1];
  const float* out_proj_w = (const float*)d_in[23];
  float* out = (float*)d_out;

  float* ws = (float*)d_ws;
  size_t off = 0;
  float* xz      = ws + off; off += (size_t)BL * NE;        // [b][e][l]
  float* convout = ws + off; off += (size_t)BL * DI;        // [b][j][d]
  float* zout    = ws + off; off += (size_t)BL * DI;        // [b][j][d]
  float* xdbl    = ws + off; off += (size_t)BL * NX;        // [bl][80]
  float* dtbuf   = ws + off; off += (size_t)BL * DI;        // [bl][DI] delta
  float* ysum    = ws + off; off += (size_t)BL * DI;        // [b][l][d]
  float* Aprod   = ws + off; off += (size_t)BATCH * CH * DS * DI;
  float* Hend    = ws + off; off += (size_t)BATCH * CH * DS * DI;
  float* Hinit   = ws + off; off += (size_t)BATCH * CH * DS * DI;
  (void)ws_size; (void)in_sizes; (void)n_in; (void)out_size;

  dim3 gblk(128);  // 4 waves/block, one 16x16 tile per wave

  // 1) in_proj: xz[b][e][l] = hidden @ in_proj_w^T   (M=5120, N=3072, K=768)
  wmma_gemm_kernel<DM, 0, 1><<<dim3((NE/16 + 3)/4, BL/16), gblk, 0, stream>>>(
      hidden, DM, in_proj_w, xz, nullptr, NE, 0);

  for (int br = 0; br < 3; ++br) {
    const float* cw   = (const float*)d_in[2 + 7*br];
    const float* cb   = (const float*)d_in[3 + 7*br];
    const float* xpw  = (const float*)d_in[4 + 7*br];
    const float* dtw  = (const float*)d_in[5 + 7*br];
    const float* dtb  = (const float*)d_in[6 + 7*br];
    const float* alog = (const float*)d_in[7 + 7*br];
    const float* Dp   = (const float*)d_in[8 + 7*br];

    // 2) depthwise conv + silu, and silu(z), in scan layout
    conv_silu_kernel<<<(BATCH*SEQ*DI + 255)/256, 256, 0, stream>>>(
        xz, cw, cb, convout, zout, br);

    // 3) x_proj: xdbl = conv @ x_proj_w^T   (M=5120, N=80, K=1536)
    wmma_gemm_kernel<DI, 0, 0><<<dim3((NX/16 + 3)/4, BL/16), gblk, 0, stream>>>(
        convout, DI, xpw, xdbl, nullptr, NX, NX);

    // 4) dt_proj + bias + softplus: delta (M=5120, N=1536, K=48)
    wmma_gemm_kernel<DR, 1, 0><<<dim3((DI/16 + 3)/4, BL/16), gblk, 0, stream>>>(
        xdbl, NX, dtw, dtbuf, dtb, DI, DI);

    // 5) chunked selective scan
    scan_pass1<<<dim3(DI/256, CH, BATCH), 256, 0, stream>>>(
        dtbuf, convout, xdbl, alog, Aprod, Hend);
    scan_pass2<<<(BATCH*DS*DI + 255)/256, 256, 0, stream>>>(Aprod, Hend, Hinit);
    scan_pass3<<<dim3(DI/256, CH, BATCH), 256, 0, stream>>>(
        dtbuf, convout, zout, xdbl, alog, Dp, Hinit, ysum, br, br > 0);
  }

  // 6) out_proj: out[b][l][o] = ysum @ out_proj_w^T  (M=5120, N=768, K=1536)
  wmma_gemm_kernel<DI, 0, 0><<<dim3((DM/16 + 3)/4, BL/16), gblk, 0, stream>>>(
      ysum, DI, out_proj_w, out, nullptr, DM, DM);
}
